// RNN_19370302505472
// MI455X (gfx1250) — matrix operand order repaired and checked by reading, not compiled, not run on hardware
//
#include <hip/hip_runtime.h>
#include <hip/hip_bf16.h>
#include <stdint.h>

// Problem dims (fixed by reference)
#define B_  64
#define C_  128
#define H_  256
#define L_  1024
#define G_  1024            // 4*H
#define NW  395264          // per-batch weight element count = 4H(C+H+2)
#define S0  131072          // G*C
#define S1  393216          // S0 + G*H
#define S2  394240          // S1 + G

typedef __attribute__((ext_vector_type(16))) __bf16 v16bf;
typedef __attribute__((ext_vector_type(8)))  float  v8f;
typedef unsigned int v4u __attribute__((ext_vector_type(4)));
typedef int          v4i __attribute__((ext_vector_type(4)));
typedef int          v8i __attribute__((ext_vector_type(8)));

union BF16x16 { v16bf v; uint4 q[2]; };

static __device__ __forceinline__ float bf_lo(unsigned u) { return __uint_as_float(u << 16); }
static __device__ __forceinline__ float bf_hi(unsigned u) { return __uint_as_float(u & 0xffff0000u); }

// ---------------- prep: convert W_ih|W_hh region to bf16 (layout preserved) ----------------
__global__ void k_convert_weights(const float* __restrict__ w, __bf16* __restrict__ wbf) {
  size_t idx = (size_t)blockIdx.x * blockDim.x + threadIdx.x;
  if (idx >= (size_t)B_ * S1) return;
  size_t b = idx / S1, rix = idx - b * S1;
  wbf[idx] = (__bf16)w[b * (size_t)NW + rix];
}

// ---------------- prep: fused bias = w[s1:s2] + w[s2:s2+G] ----------------
__global__ void k_bias(const float* __restrict__ w, float* __restrict__ bias) {
  int idx = blockIdx.x * blockDim.x + threadIdx.x;
  if (idx >= B_ * G_) return;
  int b = idx >> 10, g = idx & 1023;
  bias[idx] = w[(size_t)b * NW + S1 + g] + w[(size_t)b * NW + S2 + g];
}

// ---------------- prep: x (B,C,L) f32 -> xT (B,L,C) bf16 ----------------
__global__ void k_transpose_x(const float* __restrict__ x, __bf16* __restrict__ xT) {
  size_t idx = (size_t)blockIdx.x * blockDim.x + threadIdx.x;
  if (idx >= (size_t)B_ * C_ * L_) return;
  size_t b = idx / ((size_t)C_ * L_);
  int rem = (int)(idx - b * (size_t)C_ * L_);
  int c = rem / L_, l = rem - c * L_;
  xT[(b * L_ + l) * C_ + c] = (__bf16)x[idx];   // coalesced read, scattered b16 write (one-time)
}

// ---------------- phase 1: XG[b][l][g] = xT[b] @ W_ih[b]^T + bias (WMMA bf16) ----------------
// grid: (8*16, B); block 256 (8 waves). Wave computes 16(M=L) x 64(N=G) strip.
__global__ __launch_bounds__(256) void k_xgemm(const __bf16* __restrict__ xT,
                                               const __bf16* __restrict__ wbf,
                                               const float*  __restrict__ bias,
                                               float*        __restrict__ XG) {
  const int b    = blockIdx.y;
  const int wave = threadIdx.x >> 5;
  const int lane = threadIdx.x & 31;
  const int half = lane >> 4;      // lane half selects K sub-range per ISA layout
  const int l16  = lane & 15;
  const int m0   = (blockIdx.x >> 4) * 128 + wave * 16;   // L tile
  const int n0   = (blockIdx.x & 15) * 64;                // G tile

  const __bf16* A   = xT + ((size_t)b * L_ + (m0 + l16)) * C_;  // row m = l16 of this tile
  const __bf16* Wih = wbf + (size_t)b * S1;                     // [g][c] row-major

  v8f acc[4];
#pragma unroll
  for (int t = 0; t < 4; ++t)
#pragma unroll
    for (int i = 0; i < 8; ++i) acc[t][i] = 0.0f;

#pragma unroll
  for (int k0 = 0; k0 < C_; k0 += 32) {
    // A fragment 16x32: lane(l16,half): elems 0..7 = K[half*8 .. +7], 8..15 = K[16+half*8 ..]
    BF16x16 af;
    af.q[0] = *(const uint4*)(A + k0 + half * 8);
    af.q[1] = *(const uint4*)(A + k0 + 16 + half * 8);
#pragma unroll
    for (int t = 0; t < 4; ++t) {
      // B fragment 32x16: lane = column N; elems 0..7 = K[half*8 .. +7], 8..15 = K[16+half*8 ..]
      const __bf16* Bp = Wih + (size_t)(n0 + t * 16 + l16) * C_ + k0 + half * 8;
      BF16x16 bfr;
      bfr.q[0] = *(const uint4*)(Bp);
      bfr.q[1] = *(const uint4*)(Bp + 16);
      acc[t] = __builtin_amdgcn_wmma_f32_16x16x32_bf16(false, af.v, false, bfr.v,
                                                       (short)0, acc[t], false, false);
    }
  }

#pragma unroll
  for (int t = 0; t < 4; ++t) {
    const int n  = n0 + t * 16 + l16;
    const float bv = bias[b * G_ + n];
#pragma unroll
    for (int i = 0; i < 8; ++i) {
      const int m = m0 + half * 8 + i;          // D layout: vgpr i -> M=i (lanes<16) / M=8+i
      XG[((size_t)b * L_ + m) * G_ + n] = acc[t][i] + bv;
    }
  }
}

// ---------------- phase 2: LSTM recurrence ----------------
// grid (4, B): 4-WG cluster per batch; WG r owns hidden units j in [r*64, r*64+64),
// i.e. 256 gate rows {gate*256 + r*64 + jl}. W_hh slice lives in LDS as bf16 (padded rows),
// staged by the Tensor Data Mover (pad inserted by TDM pad_enable).
#define RS 264   // padded LDS row stride in bf16 elements (512B row + 16B pad -> bank spread)

static __device__ __forceinline__ void tdm_load_tile(uint64_t global_addr, unsigned lds_addr) {
  // D# Group 0: count=1 | lds_addr | global_addr[56:0] | type=2
  v4u g0;
  g0[0] = 1u;
  g0[1] = lds_addr;
  g0[2] = (unsigned)(global_addr & 0xffffffffu);
  g0[3] = (unsigned)((global_addr >> 32) & 0x01ffffffu) | (2u << 30);
  // D# Group 1: mask=0, data_size=2B, pad_enable, pad_interval=2^(6+1)=128 dw (512B),
  // pad_amount=3+1=4 dw (16B); tensor_dim0=256, tensor_dim1=64; tile 256x64; dim0_stride=256
  v8i g1;
  g1[0] = (int)((1u << 16) | (1u << 20) | (6u << 22) | (3u << 25));
  g1[1] = (int)((256u & 0xffffu) << 16);                       // tensor_dim0 lo16 @ bits63:48
  g1[2] = (int)(((256u >> 16) & 0xffffu) | ((64u & 0xffffu) << 16)); // dim0 hi | tensor_dim1 lo
  g1[3] = (int)(((64u >> 16) & 0xffffu) | (256u << 16));       // tensor_dim1 hi | tile_dim0=256
  g1[4] = (int)(64u);                                          // tile_dim1=64, tile_dim2=0
  g1[5] = (int)(256u);                                         // tensor_dim0_stride[31:0]=256
  g1[6] = 0;
  g1[7] = 0;
  v4i gz4 = {0, 0, 0, 0};                                      // 2D tile: groups 2/3 unused
#if defined(__clang_major__) && (__clang_major__ >= 23)
  v8i gz8 = {0, 0, 0, 0, 0, 0, 0, 0};
  __builtin_amdgcn_tensor_load_to_lds(g0, g1, gz4, gz4, gz8, 0);
#else
  __builtin_amdgcn_tensor_load_to_lds(g0, g1, gz4, gz4, 0);
#endif
}

__global__ __launch_bounds__(256) void k_lstm_rec(const __bf16* __restrict__ wbf,
                                                  const float*  __restrict__ XG,
                                                  const float*  __restrict__ state_in,
                                                  float*        __restrict__ hex,
                                                  float*        __restrict__ y,
                                                  float*        __restrict__ state_out) {
  extern __shared__ char smem[];
  __bf16* w    = (__bf16*)smem;                        // 256 rows * RS bf16 = 135168 B
  float*  hbuf = (float*)(smem + 256 * RS * 2);        // 256 f32: full h vector
  float*  gl   = hbuf + 256;                           // 256 f32: raw gate pre-activations

  const int r    = blockIdx.x;          // hidden slice 0..3
  const int b    = blockIdx.y;
  const int t    = threadIdx.x;         // local gate row 0..255
  const int gate = t >> 6;              // 0=i 1=f 2=g 3=o
  const int jl   = t & 63;
  const int gr   = gate * 256 + r * 64 + jl;   // global gate row in W_hh / xg

  // Stage this WG's W_hh rows into LDS via TDM: 4 tiles (one per gate), 64x256 bf16 each,
  // row padding (512B -> 528B) applied by the TDM. One wave issues; TENSORcnt guards.
  if (threadIdx.x < 32) {
    const __bf16* Whh_b = wbf + (size_t)b * S1 + S0;
#pragma unroll
    for (int g4 = 0; g4 < 4; ++g4) {
      uint64_t ga  = (uint64_t)(uintptr_t)(Whh_b + ((size_t)g4 * 256 + r * 64) * H_);
      unsigned lds = (unsigned)(g4 * 64 * (RS * 2));   // dynamic LDS segment starts at 0
      tdm_load_tile(ga, lds);
    }
    __builtin_amdgcn_s_wait_tensorcnt(0);
  }

  hbuf[t] = state_in[(size_t)b * (2 * H_) + t];
  float c = 0.0f, hlast = 0.0f;
  if (t < 64) c = state_in[(size_t)b * (2 * H_) + H_ + r * 64 + t];
  __syncthreads();   // publishes TDM-filled LDS + hbuf to all waves

  const float* xg_b = XG + (size_t)b * L_ * G_;
  float* hex0 = hex + (size_t)b * H_;
  float* hex1 = hex + (size_t)B_ * H_ + (size_t)b * H_;

  for (int l = 0; l < L_; ++l) {
    { // prefetch next step's xg element stream into cache
      int lp = (l + 1 < L_) ? (l + 1) : l;
      __builtin_prefetch(xg_b + (size_t)lp * G_ + gr, 0, 1);
    }

    // gates[gr] = xg[l][gr] + <h, W_hh[gr,:]>   (bf16 weights, f32 accumulate)
    float acc = xg_b[(size_t)l * G_ + gr];
    const uint4*  wrow = (const uint4*)(w + (size_t)t * RS);
    const float4* hv   = (const float4*)hbuf;
#pragma unroll 8
    for (int kk = 0; kk < 32; ++kk) {
      uint4  wv = wrow[kk];
      float4 h0 = hv[2 * kk], h1 = hv[2 * kk + 1];
      acc = fmaf(bf_lo(wv.x), h0.x, acc);
      acc = fmaf(bf_hi(wv.x), h0.y, acc);
      acc = fmaf(bf_lo(wv.y), h0.z, acc);
      acc = fmaf(bf_hi(wv.y), h0.w, acc);
      acc = fmaf(bf_lo(wv.z), h1.x, acc);
      acc = fmaf(bf_hi(wv.z), h1.y, acc);
      acc = fmaf(bf_lo(wv.w), h1.z, acc);
      acc = fmaf(bf_hi(wv.w), h1.w, acc);
    }
    gl[t] = acc;
    __syncthreads();

    float* hx = (l & 1) ? hex1 : hex0;   // ping-pong exchange buffer
    if (t < 64) {
      float iv = 1.0f / (1.0f + __expf(-gl[t]));
      float fv = 1.0f / (1.0f + __expf(-gl[64 + t]));
      float gv = tanhf(gl[128 + t]);
      float ov = 1.0f / (1.0f + __expf(-gl[192 + t]));
      c = fv * c + iv * gv;
      float h = ov * tanhf(c);
      hlast = h;
      int jj = r * 64 + t;
      hbuf[jj] = h;
      hx[jj]   = h;
      y[((size_t)b * H_ + jj) * L_ + l] = h;
    }
    __threadfence();          // release own h slice device-wide
    __syncthreads();          // all stores in WG done before wave0 signals
    if (threadIdx.x < 32) __builtin_amdgcn_s_cluster_barrier();  // one wave signals+waits (-3)
    __syncthreads();          // rest of WG waits on wave0
    __threadfence();          // acquire remote slices
    if ((t >> 6) != r) hbuf[t] = hx[t];
    __syncthreads();
  }

  if (t < 64) {
    int jj = r * 64 + t;
    state_out[(size_t)b * (2 * H_) + jj]      = hlast;
    state_out[(size_t)b * (2 * H_) + H_ + jj] = c;
  }
}

// ---------------- launch ----------------
extern "C" void kernel_launch(void* const* d_in, const int* in_sizes, int n_in,
                              void* d_out, int out_size, void* d_ws, size_t ws_size,
                              hipStream_t stream) {
  const float* x     = (const float*)d_in[0];   // B*C*L
  const float* state = (const float*)d_in[1];   // B*2H
  const float* wts   = (const float*)d_in[2];   // B*NW
  float* y         = (float*)d_out;                       // B*H*L
  float* state_out = y + (size_t)B_ * H_ * L_;            // B*2H

  char* ws = (char*)d_ws;
  __bf16* xT  = (__bf16*)(ws);                    //  16,777,216 B
  __bf16* wbf = (__bf16*)(ws + 16777216);         //  50,331,648 B  (W_ih|W_hh per batch)
  float*  bia = (float*)(ws + 67108864);          //     262,144 B
  float*  XG  = (float*)(ws + 67371008);          // 268,435,456 B
  float*  hex = (float*)(ws + 335806464);         //     131,072 B  (2 x B x H ping-pong)

  {
    size_t n = (size_t)B_ * S1;
    k_convert_weights<<<dim3((unsigned)((n + 255) / 256)), dim3(256), 0, stream>>>(wts, wbf);
  }
  {
    size_t n = (size_t)B_ * C_ * L_;
    k_transpose_x<<<dim3((unsigned)((n + 255) / 256)), dim3(256), 0, stream>>>(x, xT);
  }
  k_bias<<<dim3((B_ * G_) / 256), dim3(256), 0, stream>>>(wts, bia);

  k_xgemm<<<dim3(128, B_), dim3(256), 0, stream>>>(xT, wbf, bia, XG);

  const size_t smem = (size_t)256 * RS * 2 + 512 * sizeof(float);  // weights + hbuf + gl
  k_lstm_rec<<<dim3(4, B_), dim3(256), smem, stream>>>(wbf, XG, state, hex, y, state_out);
}
